// SweepGaterV4_83571473645672
// MI455X (gfx1250) — matrix operand order repaired and checked by reading, not compiled, not run on hardware
//
#include <hip/hip_runtime.h>
#include <math.h>

// ---------------- problem constants ----------------
#define BB   2
#define SSZ  4
#define CC   64
#define HWW  65536      // 256*256
#define CI_  194
#define CO_  24
#define COP  32         // CO padded to 2 WMMA M-tiles
#define ENC_ 96
#define HID_ 64

typedef __attribute__((ext_vector_type(16))) _Float16 v16h;
typedef __attribute__((ext_vector_type(8)))  float    v8f;

// ---------------- workspace layout (floats) --------
// total = 55,067,008 floats ~= 220.3 MB
#define OFF_TAD  ((size_t)0)                       // B*S*C*HW   = 33554432
#define OFF_ENC  (OFF_TAD  + (size_t)33554432)     // B*96*HW    = 12582912
#define OFF_H    (OFF_ENC  + (size_t)12582912)     // B*64*HW    =  8388608
#define OFF_HEUR (OFF_H    + (size_t)8388608)      // B*S*HW     =   524288
#define OFF_GS   (OFF_HEUR + (size_t)524288)       // S*32*64    =     8192
#define OFF_GT   (OFF_GS   + (size_t)8192)
#define OFF_CST  (OFF_GT   + (size_t)8192)         // S*32
#define OFF_ST1  (OFF_CST  + (size_t)128)          // B*32*2
#define OFF_ST2  (OFF_ST1  + (size_t)128)

__device__ __forceinline__ v8f wmma_f16(v16h a, v16h b, v8f c) {
  return __builtin_amdgcn_wmma_f32_16x16x32_f16(false, a, false, b, (short)0, c,
                                                false, false);
}

__device__ __forceinline__ float gelu_exact(float x) {
  return 0.5f * x * (1.0f + erff(x * 0.70710678118654752f));
}

// A fragment (16x32, f16) from f32 row-major W[row*ldk + k]   (ISA 7.12.2)
__device__ __forceinline__ v16h load_afrag_f32(const float* __restrict__ W,
                                               int ldk, int mbase, int kbase,
                                               int lane) {
  const int half = lane >> 4, r = lane & 15;
  const int row = mbase + r;
  v16h a;
#pragma unroll
  for (int v = 0; v < 8; ++v) {
    int k0 = half * 16 + 2 * v;
    a[2 * v]     = (_Float16)W[row * ldk + kbase + k0];
    a[2 * v + 1] = (_Float16)W[row * ldk + kbase + k0 + 1];
  }
  return a;
}

// B fragment (32x16, f16): K = channel (stride HW in memory), N = pixel
__device__ __forceinline__ v16h load_bfrag_feat(const float* __restrict__ X,
                                                int chbase, int pix, int lane) {
  const int half = lane >> 4, col = lane & 15;
  v16h b;
#pragma unroll
  for (int v = 0; v < 8; ++v) {
    int k0 = half * 16 + 2 * v;
    b[2 * v]     = (_Float16)X[(chbase + k0)     * HWW + pix + col];
    b[2 * v + 1] = (_Float16)X[(chbase + k0 + 1) * HWW + pix + col];
  }
  return b;
}

// ------------- kernel 0: fold encoder weights -------------
// GS[s,o,i] = sum_c (E_S+E_D)[s,o,c] * inv[s,c] * Aw[s,c,i]
// GT[s,o,i] = sum_c (E_T-E_D)[s,o,c] * inv[s,c] * Aw[s,c,i]
// CST[s,o]  = sum_c (E_S+E_T)[s,o,c] * inv[s,c] * (Ab[s,c]-mu[s,c]) + pos terms
__global__ void k_setup(const float* __restrict__ Aw, const float* __restrict__ Ab,
                        const float* __restrict__ mu, const float* __restrict__ sig,
                        const float* __restrict__ Ew, float* __restrict__ GS,
                        float* __restrict__ GT, float* __restrict__ CST) {
  for (int idx = threadIdx.x; idx < SSZ * COP * CC; idx += blockDim.x) {
    int i = idx & 63, o = (idx >> 6) & 31, s = idx >> 11;
    float gs = 0.f, gt = 0.f;
    if (o < CO_) {
      for (int c = 0; c < CC; ++c) {
        float inv = 1.f / fmaxf(sig[s * CC + c], 0.001f);
        const float* e = Ew + (s * CO_ + o) * CI_;
        float es = e[c], et = e[64 + c], ed = e[128 + c];
        float aw = Aw[(s * CC + c) * CC + i];
        gs += (es + ed) * inv * aw;
        gt += (et - ed) * inv * aw;
      }
    }
    GS[idx] = gs;
    GT[idx] = gt;
  }
  for (int idx = threadIdx.x; idx < SSZ * COP; idx += blockDim.x) {
    int o = idx & 31, s = idx >> 5;
    float v = 0.f;
    if (o < CO_) {
      const float* e = Ew + (s * CO_ + o) * CI_;
      for (int c = 0; c < CC; ++c) {
        float inv = 1.f / fmaxf(sig[s * CC + c], 0.001f);
        v += (e[c] + e[64 + c]) * inv * (Ab[s * CC + c] - mu[s * CC + c]);
      }
      float ang = 6.283185307179586f * ((float)s / (float)(SSZ - 1));
      v += e[192] * sinf(ang) + e[193] * cosf(ang);
    }
    CST[idx] = v;
  }
}

// ------------- kernel 1: adapter + encoder + heuristic (WMMA) -------------
__global__ __launch_bounds__(128)
void k_adapt_enc(const float* __restrict__ S_feat, const float* __restrict__ T_feat,
                 const float* __restrict__ Aw, const float* __restrict__ Ab,
                 const float* __restrict__ sig, const float* __restrict__ eprx,
                 const float* __restrict__ GS, const float* __restrict__ GT,
                 const float* __restrict__ CST, float* __restrict__ Tad,
                 float* __restrict__ Enc, float* __restrict__ Heur) {
  const int lane = threadIdx.x & 31;
  const int wave = threadIdx.x >> 5;
  const int nwav = blockDim.x >> 5;
  const int s = blockIdx.y & (SSZ - 1);
  const int b = blockIdx.y >> 2;
  const int half = lane >> 4, col = lane & 15;

  // weight fragments (loop-invariant)
  v16h Af[4][2], GSf[2][2], GTf[2][2];
#pragma unroll
  for (int m = 0; m < 4; ++m)
#pragma unroll
    for (int k = 0; k < 2; ++k)
      Af[m][k] = load_afrag_f32(Aw + s * CC * CC, CC, m * 16, k * 32, lane);
#pragma unroll
  for (int m = 0; m < 2; ++m)
#pragma unroll
    for (int k = 0; k < 2; ++k) {
      GSf[m][k] = load_afrag_f32(GS + s * COP * CC, CC, m * 16, k * 32, lane);
      GTf[m][k] = load_afrag_f32(GT + s * COP * CC, CC, m * 16, k * 32, lane);
    }

  float abias[4][8], ainv[4][8], ecst[2][8];
#pragma unroll
  for (int m = 0; m < 4; ++m)
#pragma unroll
    for (int r = 0; r < 8; ++r) {
      int ch = m * 16 + r + half * 8;
      abias[m][r] = Ab[s * CC + ch];
      ainv[m][r] = 1.f / fmaxf(sig[s * CC + ch], 0.001f);
    }
#pragma unroll
  for (int m = 0; m < 2; ++m)
#pragma unroll
    for (int r = 0; r < 8; ++r) ecst[m][r] = CST[s * COP + m * 16 + r + half * 8];
  const float eprox = eprx[s];

  const float* Tp = T_feat + (size_t)(b * SSZ + s) * CC * HWW;
  const float* Sp = S_feat + (size_t)b * CC * HWW;
  float* TadP  = Tad  + (size_t)(b * SSZ + s) * CC * HWW;
  float* EncP  = Enc  + (size_t)(b * ENC_ + s * CO_) * HWW;
  float* HeurP = Heur + (size_t)(b * SSZ + s) * HWW;

  for (int t = blockIdx.x * nwav + wave; t < HWW / 16; t += gridDim.x * nwav) {
    const int pix = t * 16;
    v16h Bt[2], Bs[2];
    Bt[0] = load_bfrag_feat(Tp, 0, pix, lane);
    Bt[1] = load_bfrag_feat(Tp, 32, pix, lane);
    Bs[0] = load_bfrag_feat(Sp, 0, pix, lane);
    Bs[1] = load_bfrag_feat(Sp, 32, pix, lane);

    v8f aT[4], aS[4], aE[2];
#pragma unroll
    for (int m = 0; m < 4; ++m) { aT[m] = {}; aS[m] = {}; }
#pragma unroll
    for (int m = 0; m < 2; ++m) aE[m] = {};
#pragma unroll
    for (int m = 0; m < 4; ++m)
#pragma unroll
      for (int k = 0; k < 2; ++k) {
        aT[m] = wmma_f16(Af[m][k], Bt[k], aT[m]);
        aS[m] = wmma_f16(Af[m][k], Bs[k], aS[m]);
      }
#pragma unroll
    for (int m = 0; m < 2; ++m)
#pragma unroll
      for (int k = 0; k < 2; ++k) {
        aE[m] = wmma_f16(GSf[m][k], Bs[k], aE[m]);
        aE[m] = wmma_f16(GTf[m][k], Bt[k], aE[m]);
      }

    // elementwise: bias add + store T_adapt, delta/proxy (biases & mu cancel)
    float psum = 0.f;
#pragma unroll
    for (int m = 0; m < 4; ++m)
#pragma unroll
      for (int r = 0; r < 8; ++r) {
        float tv = aT[m][r];
        float d = ainv[m][r] * (aS[m][r] - tv);
        psum += d * d;
        int ch = m * 16 + r + half * 8;
        TadP[ch * HWW + pix + col] = tv + abias[m][r];
      }
    float proxy = (psum + __shfl_xor(psum, 16, 32)) * (1.f / 64.f);
    if (half == 0) {
      float band = (proxy >= 0.05f && proxy <= 0.2f) ? 1.f : 0.f;
      HeurP[pix + col] = (proxy - eprox) + 0.5f * band;  // ALPHA=1, BETA=0.5
    }
#pragma unroll
    for (int m = 0; m < 2; ++m)
#pragma unroll
      for (int r = 0; r < 8; ++r) {
        int ch = m * 16 + r + half * 8;
        if (ch < CO_) EncP[ch * HWW + pix + col] = aE[m][r] + ecst[m][r];
      }
  }
}

// ------------- group-norm stats: one block per (b, group), no atomics -------
__global__ void k_gnstats(const float* __restrict__ X, float* __restrict__ st,
                          int Cn, int G) {
  const int g = blockIdx.x & 31, b = blockIdx.x >> 5;
  const float* base = X + ((size_t)b * Cn + g * G) * HWW;
  const int N = G * HWW;
  float s1 = 0.f, s2 = 0.f;
  for (int i = threadIdx.x; i < N; i += blockDim.x) {
    float v = base[i];
    s1 += v;
    s2 += v * v;
  }
  __shared__ float sh1[256], sh2[256];
  sh1[threadIdx.x] = s1;
  sh2[threadIdx.x] = s2;
  __syncthreads();
  for (int o = 128; o > 0; o >>= 1) {
    if ((int)threadIdx.x < o) {
      sh1[threadIdx.x] += sh1[threadIdx.x + o];
      sh2[threadIdx.x] += sh2[threadIdx.x + o];
    }
    __syncthreads();
  }
  if (threadIdx.x == 0) {
    float mean = sh1[0] / (float)N;
    float var = sh2[0] / (float)N - mean * mean;
    st[blockIdx.x * 2] = mean;
    st[blockIdx.x * 2 + 1] = rsqrtf(var + 1e-5f);
  }
}

// ------------- kernel 3: GN1 + GELU + mix1 (WMMA) -------------
__global__ __launch_bounds__(128)
void k_mix1(const float* __restrict__ Enc, const float* __restrict__ st1,
            const float* __restrict__ g1g, const float* __restrict__ g1b,
            const float* __restrict__ m1w, float* __restrict__ Hh) {
  const int lane = threadIdx.x & 31;
  const int wave = threadIdx.x >> 5;
  const int nwav = blockDim.x >> 5;
  const int b = blockIdx.y;
  const int half = lane >> 4, col = lane & 15;

  v16h Af[4][3];
#pragma unroll
  for (int m = 0; m < 4; ++m)
#pragma unroll
    for (int k = 0; k < 3; ++k)
      Af[m][k] = load_afrag_f32(m1w, ENC_, m * 16, k * 32, lane);

  float scl[3][16], sft[3][16];
#pragma unroll
  for (int k = 0; k < 3; ++k)
#pragma unroll
    for (int j = 0; j < 16; ++j) {
      int ch = k * 32 + half * 16 + j;
      int grp = ch / 3;
      float mean = st1[(b * 32 + grp) * 2], istd = st1[(b * 32 + grp) * 2 + 1];
      float ga = g1g[ch];
      scl[k][j] = ga * istd;
      sft[k][j] = g1b[ch] - mean * ga * istd;
    }

  const float* Eb = Enc + (size_t)b * ENC_ * HWW;
  float* Hb = Hh + (size_t)b * HID_ * HWW;

  for (int t = blockIdx.x * nwav + wave; t < HWW / 16; t += gridDim.x * nwav) {
    const int pix = t * 16;
    v16h Bf[3];
#pragma unroll
    for (int k = 0; k < 3; ++k)
#pragma unroll
      for (int j = 0; j < 16; ++j) {
        int ch = k * 32 + half * 16 + j;
        float e = Eb[ch * HWW + pix + col];
        Bf[k][j] = (_Float16)gelu_exact(e * scl[k][j] + sft[k][j]);
      }
    v8f acc[4];
#pragma unroll
    for (int m = 0; m < 4; ++m) acc[m] = {};
#pragma unroll
    for (int m = 0; m < 4; ++m)
#pragma unroll
      for (int k = 0; k < 3; ++k) acc[m] = wmma_f16(Af[m][k], Bf[k], acc[m]);
#pragma unroll
    for (int m = 0; m < 4; ++m)
#pragma unroll
      for (int r = 0; r < 8; ++r)
        Hb[(m * 16 + r + half * 8) * HWW + pix + col] = acc[m][r];
  }
}

// ------------- kernel 5: GN2 + GELU + mix2 + softmax gating -------------
__global__ void k_final(const float* __restrict__ Hh, const float* __restrict__ st2,
                        const float* __restrict__ g2g, const float* __restrict__ g2b,
                        const float* __restrict__ m2w, const float* __restrict__ m2b,
                        const float* __restrict__ Heur, const float* __restrict__ Tad,
                        float* __restrict__ out) {
  int p = blockIdx.x * blockDim.x + threadIdx.x;
  if (p >= BB * HWW) return;
  const int b = p / HWW, pix = p - b * HWW;
  const float* Hb = Hh + (size_t)b * HID_ * HWW + pix;

  float lear[SSZ] = {0.f, 0.f, 0.f, 0.f};
  for (int c = 0; c < HID_; ++c) {
    int grp = c >> 1;
    float mean = st2[(b * 32 + grp) * 2], istd = st2[(b * 32 + grp) * 2 + 1];
    float y = g2g[c] * (Hb[c * HWW] - mean) * istd + g2b[c];
    float g = gelu_exact(y);
#pragma unroll
    for (int s = 0; s < SSZ; ++s) lear[s] += m2w[s * HID_ + c] * g;
  }
  float lg[SSZ], mx = -1e30f;
#pragma unroll
  for (int s = 0; s < SSZ; ++s) {
    float h = Heur[((size_t)(b * SSZ + s)) * HWW + pix];
    lg[s] = (0.5f * h + 0.5f * (lear[s] + m2b[s])) * (1.f / 0.7f);
    mx = fmaxf(mx, lg[s]);
  }
  float gate[SSZ], sum = 0.f;
#pragma unroll
  for (int s = 0; s < SSZ; ++s) { gate[s] = __expf(lg[s] - mx); sum += gate[s]; }
  float rs = 1.f / sum;
#pragma unroll
  for (int s = 0; s < SSZ; ++s) gate[s] *= rs;

  const float* Tb = Tad + (size_t)b * SSZ * CC * HWW + pix;
  float* Ob = out + (size_t)b * CC * HWW + pix;
  for (int c = 0; c < CC; ++c) {
    float acc = 0.f;
#pragma unroll
    for (int s = 0; s < SSZ; ++s) acc += gate[s] * Tb[(s * CC + c) * HWW];
    Ob[c * HWW] = acc;
  }
}

// ----------------------- host entry -----------------------
extern "C" void kernel_launch(void* const* d_in, const int* in_sizes, int n_in,
                              void* d_out, int out_size, void* d_ws, size_t ws_size,
                              hipStream_t stream) {
  (void)in_sizes; (void)n_in; (void)out_size; (void)ws_size;
  const float* S_feat = (const float*)d_in[0];
  const float* T_feat = (const float*)d_in[1];
  const float* Aw     = (const float*)d_in[2];
  const float* Ab     = (const float*)d_in[3];
  const float* mu     = (const float*)d_in[4];
  const float* sig    = (const float*)d_in[5];
  const float* eprx   = (const float*)d_in[6];
  const float* Ew     = (const float*)d_in[7];
  const float* g1g    = (const float*)d_in[8];
  const float* g1b    = (const float*)d_in[9];
  const float* m1w    = (const float*)d_in[10];
  const float* g2g    = (const float*)d_in[11];
  const float* g2b    = (const float*)d_in[12];
  const float* m2w    = (const float*)d_in[13];
  const float* m2b    = (const float*)d_in[14];
  float* ws = (float*)d_ws;
  float* Tad  = ws + OFF_TAD;
  float* Enc  = ws + OFF_ENC;
  float* Hh   = ws + OFF_H;
  float* Heur = ws + OFF_HEUR;
  float* GS   = ws + OFF_GS;
  float* GT   = ws + OFF_GT;
  float* CST  = ws + OFF_CST;
  float* st1  = ws + OFF_ST1;
  float* st2  = ws + OFF_ST2;
  float* out  = (float*)d_out;

  k_setup<<<1, 256, 0, stream>>>(Aw, Ab, mu, sig, Ew, GS, GT, CST);
  k_adapt_enc<<<dim3(128, BB * SSZ), 128, 0, stream>>>(
      S_feat, T_feat, Aw, Ab, sig, eprx, GS, GT, CST, Tad, Enc, Heur);
  k_gnstats<<<BB * 32, 256, 0, stream>>>(Enc, st1, ENC_, 3);
  k_mix1<<<dim3(128, BB), 128, 0, stream>>>(Enc, st1, g1g, g1b, m1w, Hh);
  k_gnstats<<<BB * 32, 256, 0, stream>>>(Hh, st2, HID_, 2);
  k_final<<<(BB * HWW + 255) / 256, 256, 0, stream>>>(Hh, st2, g2g, g2b, m2w,
                                                      m2b, Heur, Tad, out);
}
